// SynthesizedAttention_33792802685562
// MI455X (gfx1250) — matrix operand order repaired and checked by reading, not compiled, not run on hardware
//
#include <hip/hip_runtime.h>

typedef __attribute__((ext_vector_type(16))) _Float16 v16h;
typedef __attribute__((ext_vector_type(8)))  _Float16 v8h;
typedef __attribute__((ext_vector_type(4)))  _Float16 v4h;
typedef __attribute__((ext_vector_type(8)))  float    v8f;
typedef __attribute__((ext_vector_type(4)))  float    f32x4;

#define BATCH   4
#define HEADS   16
#define SEQ     2048
#define DH      64
#define DM      1024
#define KC      32                 // k-chunk per WMMA step
#define NCHUNK  (SEQ / KC)         // 64
#define WAVES   8                  // waves per workgroup (wave32)
#define LDSTR   40                 // padded LDS row stride in halfs (32 + 8)

// ---------------------------------------------------------------------------
// Stage 1: fused online-softmax + p@v, one streaming pass over B (1.07 GB).
// grid = (SEQ/128, BATCH*HEADS), block = 256 (8 waves); wave -> 16 q rows.
// Writes permuted attention output as f16 into workspace:
//   X[(b*SEQ + q)*DM + d*HEADS + h]
// ---------------------------------------------------------------------------
__global__ __launch_bounds__(256)
void attn_softmax_pv_kernel(const float* __restrict__ Blog,
                            const float* __restrict__ V,
                            _Float16*    __restrict__ Xo) {
  __shared__ __align__(16) _Float16 lv[2][DH][LDSTR];   // v chunk, transposed [d][k]

  const int tid  = threadIdx.x;
  const int wave = tid >> 5;
  const int lane = tid & 31;
  const int lrow = lane & 15;          // row / column index within tile
  const int hi   = lane >> 4;          // lane-pair half (0/1)

  const int bh = blockIdx.y;           // b*HEADS + h
  const int q0 = blockIdx.x * (WAVES * 16) + wave * 16;

  // per-lane logits base: row q0+lrow, K-run start offset hi*8
  const float* Brow  = Blog + ((size_t)bh * SEQ + (q0 + lrow)) * SEQ + hi * 8;
  // cooperative v-chunk load mapping: thread -> (k = tid/8, d = (tid%8)*8 .. +7)
  const int    vk    = tid >> 3;
  const int    vd    = (tid & 7) * 8;
  const float* Vbase = V + (size_t)bh * SEQ * DH;

  v8f   acc[4] = {};                   // 16x64 f32 accumulator (4 WMMA C tiles)
  float m_run  = -1e30f;               // running row max   (dup in lane pair)
  float l_run  = 0.0f;                 // running row denom (dup in lane pair)

  for (int c = 0; c < NCHUNK; ++c) {
    const int k0  = c * KC;
    const int buf = c & 1;

    // ---- cooperative: v[k0+vk][vd..vd+7] f32 -> transposed f16 in LDS ----
    {
      const float* vp = Vbase + (size_t)(k0 + vk) * DH + vd;
      f32x4 a = *(const f32x4*)vp;
      f32x4 b = *(const f32x4*)(vp + 4);
#pragma unroll
      for (int i = 0; i < 4; ++i) lv[buf][vd + i][vk]     = (_Float16)a[i];
#pragma unroll
      for (int i = 0; i < 4; ++i) lv[buf][vd + 4 + i][vk] = (_Float16)b[i];
    }

    // ---- per-lane logits: 16 f32 (two runs of 8, A-fragment order) ----
    f32x4 x0 = *(const f32x4*)(Brow + k0);
    f32x4 x1 = *(const f32x4*)(Brow + k0 + 4);
    f32x4 x2 = *(const f32x4*)(Brow + k0 + 16);
    f32x4 x3 = *(const f32x4*)(Brow + k0 + 20);
    if (c + 1 < NCHUNK) {                          // stream-ahead hint
      __builtin_prefetch(Brow + k0 + KC, 0, 3);    // -> global_prefetch_b8
      __builtin_prefetch(Brow + k0 + KC + 16, 0, 3);
    }

    float xv[16];
#pragma unroll
    for (int i = 0; i < 4; ++i) {
      xv[i] = x0[i]; xv[4 + i] = x1[i]; xv[8 + i] = x2[i]; xv[12 + i] = x3[i];
    }

    // ---- online softmax update (row state duplicated in lane pair) ----
    float mx = xv[0];
#pragma unroll
    for (int i = 1; i < 16; ++i) mx = fmaxf(mx, xv[i]);
    mx = fmaxf(mx, __shfl_xor(mx, 16, 32));        // combine lane pair -> row max
    const float m_new = fmaxf(m_run, mx);
    const float resc  = __expf(m_run - m_new);

    v16h  afrag;
    float psum = 0.0f;
#pragma unroll
    for (int i = 0; i < 16; ++i) {
      float ev = __expf(xv[i] - m_new);
      psum += ev;
      afrag[i] = (_Float16)ev;
    }
    psum += __shfl_xor(psum, 16, 32);              // full row partial sum
    l_run = l_run * resc + psum;
    m_run = m_new;

    // broadcast per-row rescale factor to C layout rows (M = r + 8*hi)
    float s[8];
#pragma unroll
    for (int r = 0; r < 8; ++r) s[r] = __shfl(resc, r + 8 * hi, 32);
#pragma unroll
    for (int r = 0; r < 8; ++r) {
      acc[0][r] *= s[r]; acc[1][r] *= s[r]; acc[2][r] *= s[r]; acc[3][r] *= s[r];
    }

    __syncthreads();                               // LDS v chunk ready

    // ---- B fragments from LDS + 4x WMMA ----
#pragma unroll
    for (int t = 0; t < 4; ++t) {
      const _Float16* bp = &lv[buf][t * 16 + lrow][hi * 8];
      v8h b0 = *(const v8h*)bp;
      v8h b1 = *(const v8h*)(bp + 16);
      v16h bfrag;
#pragma unroll
      for (int i = 0; i < 8; ++i) { bfrag[i] = b0[i]; bfrag[8 + i] = b1[i]; }
      acc[t] = __builtin_amdgcn_wmma_f32_16x16x32_f16(
          false, afrag, false, bfrag, (short)0, acc[t], false, false);
    }
  }

  // ---- normalize by row denominator and store permuted f16 output ----
  const float inv = 1.0f / l_run;
  float si[8];
#pragma unroll
  for (int r = 0; r < 8; ++r) si[r] = __shfl(inv, r + 8 * hi, 32);

  const int  h  = bh & (HEADS - 1);
  const int  bb = bh >> 4;
#pragma unroll
  for (int t = 0; t < 4; ++t) {
    const int d   = t * 16 + lrow;
    const int col = d * HEADS + h;
#pragma unroll
    for (int r = 0; r < 8; ++r) {
      const int row = q0 + r + 8 * hi;
      Xo[((size_t)bb * SEQ + row) * DM + col] = (_Float16)(acc[t][r] * si[r]);
    }
  }
}

// ---------------------------------------------------------------------------
// Stage 2a: W f32 -> f16 (2 MB, L2-resident for the GEMM)
// ---------------------------------------------------------------------------
__global__ __launch_bounds__(256)
void w_to_f16_kernel(const float* __restrict__ W, _Float16* __restrict__ Wh, int n) {
  const int i = (blockIdx.x * blockDim.x + threadIdx.x) * 4;
  if (i < n) {
    f32x4 a = *(const f32x4*)(W + i);
    v4h o;
#pragma unroll
    for (int j = 0; j < 4; ++j) o[j] = (_Float16)a[j];
    *(v4h*)(Wh + i) = o;
  }
}

// ---------------------------------------------------------------------------
// Stage 2b: Y[8192,1024] = X @ W^T + bias, WMMA f16 -> f32.
// grid = (DM/16, 8192/128), block = 256; wave -> one 16x16 output tile.
// ---------------------------------------------------------------------------
__global__ __launch_bounds__(256)
void linear_gemm_kernel(const _Float16* __restrict__ X,
                        const _Float16* __restrict__ Wh,
                        const float*    __restrict__ bias,
                        float*          __restrict__ Y) {
  const int wave = threadIdx.x >> 5;
  const int lane = threadIdx.x & 31;
  const int lrow = lane & 15;
  const int hi   = lane >> 4;

  const int i0 = blockIdx.y * (WAVES * 16) + wave * 16;
  const int j0 = blockIdx.x * 16;

  v8f acc;
  const float bj = bias[j0 + lrow];
#pragma unroll
  for (int r = 0; r < 8; ++r) acc[r] = bj;       // fold bias into accumulator

  const _Float16* xrow = X  + (size_t)(i0 + lrow) * DM + hi * 8;   // A-frag base
  const _Float16* wrow = Wh + (size_t)(j0 + lrow) * DM + hi * 8;   // B-frag base

  for (int kc = 0; kc < DM; kc += KC) {
    v8h a0 = *(const v8h*)(xrow + kc);
    v8h a1 = *(const v8h*)(xrow + kc + 16);
    v8h b0 = *(const v8h*)(wrow + kc);
    v8h b1 = *(const v8h*)(wrow + kc + 16);
    v16h afrag, bfrag;
#pragma unroll
    for (int i = 0; i < 8; ++i) {
      afrag[i] = a0[i]; afrag[8 + i] = a1[i];
      bfrag[i] = b0[i]; bfrag[8 + i] = b1[i];
    }
    acc = __builtin_amdgcn_wmma_f32_16x16x32_f16(
        false, afrag, false, bfrag, (short)0, acc, false, false);
  }

#pragma unroll
  for (int r = 0; r < 8; ++r)
    Y[(size_t)(i0 + r + 8 * hi) * DM + j0 + lrow] = acc[r];
}

// ---------------------------------------------------------------------------
extern "C" void kernel_launch(void* const* d_in, const int* in_sizes, int n_in,
                              void* d_out, int out_size, void* d_ws, size_t ws_size,
                              hipStream_t stream) {
  (void)in_sizes; (void)n_in; (void)out_size; (void)ws_size;

  const float* Blog = (const float*)d_in[0];   // [4,16,2048,2048] f32
  const float* V    = (const float*)d_in[1];   // [4,16,2048,64]   f32
  const float* W    = (const float*)d_in[2];   // [1024,1024]      f32
  const float* bias = (const float*)d_in[3];   // [1024]           f32
  float*       out  = (float*)d_out;           // [4,2048,1024]    f32

  _Float16* Xh = (_Float16*)d_ws;                       // 16 MB: permuted attn (f16)
  _Float16* Wh = Xh + (size_t)BATCH * SEQ * DM;         //  2 MB: W in f16

  attn_softmax_pv_kernel<<<dim3(SEQ / (WAVES * 16), BATCH * HEADS), 256, 0, stream>>>(
      Blog, V, Xh);
  w_to_f16_kernel<<<dim3((DM * DM) / (256 * 4)), 256, 0, stream>>>(W, Wh, DM * DM);
  linear_gemm_kernel<<<dim3(DM / 16, (BATCH * SEQ) / (WAVES * 16)), 256, 0, stream>>>(
      Xh, Wh, bias, out);
}
